// T5MultiHeadAttention_23270132810085
// MI455X (gfx1250) — matrix operand order repaired and checked by reading, not compiled, not run on hardware
//
#include <hip/hip_runtime.h>
#include <hip/hip_bf16.h>

// Problem constants (from reference): B=2, S=2048, D_FF=1024, H=16, DK=64, INNER=1024
#define B_      2
#define S_      2048
#define DFF_    1024
#define NH_     16
#define DK_     64
#define INNER_  1024

typedef __bf16 v16bf __attribute__((ext_vector_type(16)));
typedef __bf16 v8bf  __attribute__((ext_vector_type(8)));
typedef float  v8f   __attribute__((ext_vector_type(8)));
typedef int    gv4i  __attribute__((vector_size(16)));   // matches builtin param

// ---------- async global->LDS copy (ASYNCcnt path), guarded ----------

#if defined(__has_builtin)
#  if __has_builtin(__builtin_amdgcn_global_load_async_to_lds_b128)
#    define HAVE_ASYNC_LDS 1
#  endif
#endif
#ifndef HAVE_ASYNC_LDS
#  define HAVE_ASYNC_LDS 0
#endif

static __device__ __forceinline__ void async_copy16(const void* g, void* l) {
#if HAVE_ASYNC_LDS
  __builtin_amdgcn_global_load_async_to_lds_b128(
      (__attribute__((address_space(1))) gv4i*)g,
      (__attribute__((address_space(3))) gv4i*)l, 0, 0);
#else
  *(uint4*)l = *(const uint4*)g;
#endif
}

static __device__ __forceinline__ void async_wait_all() {
#if HAVE_ASYNC_LDS && __has_builtin(__builtin_amdgcn_s_wait_asynccnt)
  __builtin_amdgcn_s_wait_asynccnt(0);
#endif
}

// ---------- helpers ----------

static __device__ __forceinline__ unsigned short f32_to_bf16_bits(float f) {
  union { float f; unsigned u; } u_; u_.f = f;
  unsigned r = u_.u + 0x7fffu + ((u_.u >> 16) & 1u);   // round-to-nearest-even
  return (unsigned short)(r >> 16);
}
static __device__ __forceinline__ __bf16 f2bf(float f) {
  unsigned short b = f32_to_bf16_bits(f);
  return __builtin_bit_cast(__bf16, b);
}

// A-matrix fragment, 16x32 bf16 (M x K), row-major source with leading dim ld.
// Lanes 0-15 hold M=lane, K=c+0..15;
//                 lanes 16-31 hold M=lane-16, K=c+16..31.
static __device__ __forceinline__ v16bf load_a_frag(const __bf16* __restrict__ base,
                                                    int ld, int row, int kchunk, int hl) {
  const __bf16* p = base + (size_t)row * ld + kchunk + hl * 16;
  v8bf lo = *(const v8bf*)p;
  v8bf hi = *(const v8bf*)(p + 8);
  v16bf r;
  #pragma unroll
  for (int i = 0; i < 8; ++i) { r[i] = lo[i]; r[8 + i] = hi[i]; }
  return r;
}

// B-matrix fragment, 32x16 bf16 (K x N), from a row-major [N, K] source
// (i.e. B = source^T). Lane holds column N=lane%16 and 16 contiguous K values
// (lanes 0-15: K=c+0..15, lanes 16-31: K=c+16..31).
static __device__ __forceinline__ v16bf load_b_frag(const __bf16* __restrict__ base,
                                                    int ld, int col, int kchunk, int hl) {
  const __bf16* p = base + (size_t)col * ld + kchunk + hl * 16;
  v8bf lo = *(const v8bf*)p;
  v8bf hi = *(const v8bf*)(p + 8);
  v16bf r;
  #pragma unroll
  for (int i = 0; i < 8; ++i) { r[i] = lo[i]; r[8 + i] = hi[i]; }
  return r;
}

static __device__ __forceinline__ v8f wmma_bf(v16bf a, v16bf b, v8f c) {
  // D = A x B + C ; v_wmma_f32_16x16x32_bf16
  return __builtin_amdgcn_wmma_f32_16x16x32_bf16(false, a, false, b, (short)0, c, false, false);
}

// ---------- kernel 1: f32 -> bf16 convert (vectorized x4) ----------

__global__ __launch_bounds__(256)
void cvt_f32_bf16(const float* __restrict__ src, __bf16* __restrict__ dst, int n4) {
  int i = blockIdx.x * 256 + threadIdx.x;
  if (i < n4) {
    float4 f = ((const float4*)src)[i];
    ushort4 u;
    u.x = f32_to_bf16_bits(f.x);
    u.y = f32_to_bf16_bits(f.y);
    u.z = f32_to_bf16_bits(f.z);
    u.w = f32_to_bf16_bits(f.w);
    ((ushort4*)dst)[i] = u;
  }
}

// ---------- kernel 2: blocked GEMM  Y[m,n] = sum_k X[m,k] * W[n,k] ----------
// Block tile: M=128 (8 waves x 16 rows), N=64, K-step 32.
// W tile (64x32 bf16, 4KB) staged in double-buffered LDS via async copies and
// shared by all 8 waves; each wave computes 16x64 (4 C tiles, A reused 4x).
// mode 0: write Qh[b,h,s,dk]   (bf16)
// mode 1: write Kh[b,h,s,dk]   (bf16)
// mode 2: write Vt[b,h,dk,s]   (bf16, transposed for the ctx WMMA A operand)
// mode 3: write fout[m,n]      (f32, final output projection)
#define PROJ_BM 128
#define PROJ_BN 64
#define PROJ_BK 32

__global__ __launch_bounds__(256)
void proj_gemm_kernel(const __bf16* __restrict__ X, const __bf16* __restrict__ W,
                      __bf16* __restrict__ hout, float* __restrict__ fout, int mode) {
  __shared__ __bf16 ldsW[2][PROJ_BN * PROJ_BK];   // 2 x 4KB

  const int tid  = threadIdx.x;
  const int lid  = tid & 31;
  const int wave = tid >> 5;
  const int hl   = lid >> 4;
  const int l16  = lid & 15;

  const int bm = blockIdx.x / (INNER_ / PROJ_BN);   // 32 m-blocks
  const int bn = blockIdx.x % (INNER_ / PROJ_BN);   // 16 n-blocks
  const int mBase = bm * PROJ_BM + wave * 16;
  const int nBase = bn * PROJ_BN;

  // staging assignment: 256 threads x 16B = 4KB
  const int srow = tid >> 2;                        // 0..63 : n within tile
  const int soff = (tid & 3) * 8;                   // k offset (elems)
  const __bf16* wsrc = W + (size_t)(nBase + srow) * DFF_ + soff;
  const int     wdst = srow * PROJ_BK + soff;

  v8f acc[4] = {{}, {}, {}, {}};

  async_copy16(wsrc, &ldsW[0][wdst]);               // prologue: stage k-step 0

  const int NK = DFF_ / PROJ_BK;                    // 32
  for (int i = 0; i < NK; ++i) {
    const int cur = i & 1;
    async_wait_all();
    __syncthreads();
    if (i + 1 < NK)
      async_copy16(wsrc + (i + 1) * PROJ_BK, &ldsW[cur ^ 1][wdst]);

    v16bf a = load_a_frag(X, DFF_, mBase + l16, i * PROJ_BK, hl);
    const __bf16* lb = &ldsW[cur][0];
    #pragma unroll
    for (int j = 0; j < 4; ++j) {
      v16bf b = load_b_frag(lb, PROJ_BK, j * 16 + l16, 0, hl);
      acc[j] = wmma_bf(a, b, acc[j]);
    }
  }

  // C/D layout: VGPR r, lanes 0-15: M=r; lanes 16-31: M=r+8; N=lane%16
  #pragma unroll
  for (int j = 0; j < 4; ++j) {
    const int nn = nBase + j * 16 + l16;
    if (mode == 3) {
      #pragma unroll
      for (int r = 0; r < 8; ++r) {
        int mm = mBase + r + hl * 8;
        fout[(size_t)mm * INNER_ + nn] = acc[j][r];
      }
    } else {
      const int hh = nn >> 6;          // head
      const int dd = nn & (DK_ - 1);   // dim-in-head
      #pragma unroll
      for (int r = 0; r < 8; ++r) {
        int mm = mBase + r + hl * 8;
        int bi = mm >> 11;             // / S_
        int ss = mm & (S_ - 1);
        __bf16 val = f2bf(acc[j][r]);
        if (mode == 2)
          hout[(((size_t)bi * NH_ + hh) * DK_ + dd) * S_ + ss] = val;    // V^T
        else
          hout[(((size_t)bi * NH_ + hh) * S_ + ss) * DK_ + dd] = val;    // Qh / Kh
      }
    }
  }
}

// ---------- kernel 3: fused flash attention ----------
// Block = 8 waves on one (b,h); wave w owns query tile qblk*8+w (16 queries).
// Each 32-key step: K tile (32x64) and V^T tile (64x32) staged once in
// double-buffered LDS (async) and shared by all 8 waves.
// Computes S^T = Kh * Qh^T (softmax stats per lane column), online softmax
// with additive mask, ctx^T += V^T * P^T via WMMA.
#define ATT_KB 32

__global__ __launch_bounds__(256)
void attn_kernel(const __bf16* __restrict__ Qh, const __bf16* __restrict__ Kh,
                 const __bf16* __restrict__ Vt, const float* __restrict__ mask,
                 __bf16* __restrict__ ctxb) {
  __shared__ __bf16 ldsK[2][ATT_KB * DK_];   // [sk][d], 2 x 4KB
  __shared__ __bf16 ldsV[2][DK_ * ATT_KB];   // [d][sk], 2 x 4KB

  const int tid  = threadIdx.x;
  const int lid  = tid & 31;
  const int wave = tid >> 5;
  const int hl   = lid >> 4;
  const int l16  = lid & 15;

  const int bh   = blockIdx.x >> 4;          // 0..31
  const int qblk = blockIdx.x & 15;          // 16 q-blocks of 128 queries
  const int bi   = bh >> 4;
  const int hh   = bh & 15;
  const int qt   = qblk * 8 + wave;          // query tile (16 rows)
  const int sqBase = qt * 16;
  const int sq     = sqBase + l16;

  const __bf16* Qp = Qh + (size_t)bh * S_ * DK_;
  const __bf16* Kp = Kh + (size_t)bh * S_ * DK_;
  const __bf16* Vp = Vt + (size_t)bh * DK_ * S_;
  const float*  Mp = mask + (size_t)bi * S_ * S_;

  // staging: K tile 4KB (row=sk, 64 elems) ; V tile 4KB (row=d, 32 elems)
  const __bf16* ksrc = Kp + (size_t)(tid >> 3) * DK_ + (tid & 7) * 8;
  const __bf16* vsrc = Vp + (size_t)(tid >> 2) * S_ + (tid & 3) * 8;
  const int     kdst = (tid >> 3) * DK_ + (tid & 7) * 8;
  const int     vdst = (tid >> 2) * ATT_KB + (tid & 3) * 8;

  // Q^T as B-fragments (column sq = lane%16, K = d), loaded once per wave.
  v16bf qb0 = load_b_frag(Qp, DK_, sq, 0, hl);
  v16bf qb1 = load_b_frag(Qp, DK_, sq, 32, hl);

  float mrow = -3.0e38f, lrow = 0.0f;
  v8f c0 = {}, c1 = {}, c2 = {}, c3 = {};     // ctx^T accumulators, d-tiles 0..3

  // prologue: stage key block 0
  async_copy16(ksrc, &ldsK[0][kdst]);
  async_copy16(vsrc, &ldsV[0][vdst]);

  const int NKB = S_ / ATT_KB;                // 64
  for (int i = 0; i < NKB; ++i) {
    const int cur = i & 1;
    async_wait_all();
    __syncthreads();
    if (i + 1 < NKB) {
      async_copy16(ksrc + (size_t)(i + 1) * ATT_KB * DK_, &ldsK[cur ^ 1][kdst]);
      async_copy16(vsrc + (i + 1) * ATT_KB,               &ldsV[cur ^ 1][vdst]);
    }
    const int kb = i * ATT_KB;
    const __bf16* lK = &ldsK[cur][0];
    const __bf16* lV = &ldsV[cur][0];

    // ---- scores^T: two 16x16 C tiles (sk 0-15 and 16-31 of this key block)
    v8f s0 = {}, s1 = {};
    s0 = wmma_bf(load_a_frag(lK, DK_,      l16,  0, hl), qb0, s0);
    s0 = wmma_bf(load_a_frag(lK, DK_,      l16, 32, hl), qb1, s0);
    s1 = wmma_bf(load_a_frag(lK, DK_, 16 + l16,  0, hl), qb0, s1);
    s1 = wmma_bf(load_a_frag(lK, DK_, 16 + l16, 32, hl), qb1, s1);

    // ---- additive mask bias (T5: no 1/sqrt(d) scaling)
    const float* mrowp = Mp + (size_t)sq * S_ + kb + hl * 8;
    __builtin_prefetch(mrowp + ATT_KB, 0, 0);      // next key block's mask row
    #pragma unroll
    for (int r = 0; r < 8; ++r) {
      s0[r] += mrowp[r];
      s1[r] += mrowp[16 + r];
    }

    // ---- online softmax over the sk axis (per lane column sq)
    float tmax = fmaxf(s0[0], s1[0]);
    #pragma unroll
    for (int r = 1; r < 8; ++r) tmax = fmaxf(tmax, fmaxf(s0[r], s1[r]));
    tmax = fmaxf(tmax, __shfl_xor(tmax, 16, 32));
    float mnew  = fmaxf(mrow, tmax);
    float alpha = __expf(mrow - mnew);

    v8f p0, p1;
    float rsum = 0.0f;
    #pragma unroll
    for (int r = 0; r < 8; ++r) {
      p0[r] = __expf(s0[r] - mnew);
      p1[r] = __expf(s1[r] - mnew);
      rsum += p0[r] + p1[r];
    }
    rsum += __shfl_xor(rsum, 16, 32);
    lrow = lrow * alpha + rsum;
    mrow = mnew;
    #pragma unroll
    for (int r = 0; r < 8; ++r) { c0[r] *= alpha; c1[r] *= alpha; c2[r] *= alpha; c3[r] *= alpha; }

    // ---- repack P (C layout) into a B-fragment (K=sk, N=sq): one half-swap
    v16bf pb;
    #pragma unroll
    for (int r = 0; r < 8; ++r) {
      float o0 = __shfl_xor(p0[r], 16, 32);
      float o1 = __shfl_xor(p1[r], 16, 32);
      pb[r]     = f2bf(hl ? o1 : p0[r]);   // K = c+0..7
      pb[8 + r] = f2bf(hl ? p1[r] : o0);   // K = c+8..15
    }

    // ---- ctx^T += V^T_tile(16x32) x P(32x16), 4 d-tiles (from LDS)
    c0 = wmma_bf(load_a_frag(lV, ATT_KB,  0 + l16, 0, hl), pb, c0);
    c1 = wmma_bf(load_a_frag(lV, ATT_KB, 16 + l16, 0, hl), pb, c1);
    c2 = wmma_bf(load_a_frag(lV, ATT_KB, 32 + l16, 0, hl), pb, c2);
    c3 = wmma_bf(load_a_frag(lV, ATT_KB, 48 + l16, 0, hl), pb, c3);
  }

  // ---- normalize and scatter ctx back to [B,S,INNER] (bf16 for final GEMM)
  float inv = 1.0f / lrow;
  size_t rowoff = ((size_t)bi * S_ + sq) * INNER_ + hh * DK_;
  #pragma unroll
  for (int r = 0; r < 8; ++r) {
    int dbase = r + hl * 8;
    ctxb[rowoff +  0 + dbase] = f2bf(c0[r] * inv);
    ctxb[rowoff + 16 + dbase] = f2bf(c1[r] * inv);
    ctxb[rowoff + 32 + dbase] = f2bf(c2[r] * inv);
    ctxb[rowoff + 48 + dbase] = f2bf(c3[r] * inv);
  }
}

// ---------- host launch ----------

extern "C" void kernel_launch(void* const* d_in, const int* in_sizes, int n_in,
                              void* d_out, int out_size, void* d_ws, size_t ws_size,
                              hipStream_t stream) {
  (void)in_sizes; (void)n_in; (void)out_size; (void)ws_size;
  const float* q    = (const float*)d_in[0];
  const float* k    = (const float*)d_in[1];
  const float* v    = (const float*)d_in[2];
  const float* mask = (const float*)d_in[3];
  const float* Wq   = (const float*)d_in[4];
  const float* Wk   = (const float*)d_in[5];
  const float* Wv   = (const float*)d_in[6];
  const float* Wo   = (const float*)d_in[7];
  float* out = (float*)d_out;

  const size_t NA = (size_t)B_ * S_ * DFF_;      // 4M activation elems
  const size_t NW = (size_t)INNER_ * DFF_;       // 1M weight elems

  __bf16* base = (__bf16*)d_ws;
  __bf16* qbf  = base;
  __bf16* kbf  = qbf + NA;
  __bf16* vbf  = kbf + NA;
  __bf16* wqb  = vbf + NA;
  __bf16* wkb  = wqb + NW;
  __bf16* wvb  = wkb + NW;
  __bf16* wob  = wvb + NW;
  __bf16* Qhb  = wob + NW;
  __bf16* Khb  = Qhb + NA;
  __bf16* Vtb  = Khb + NA;
  __bf16* ctxb = Vtb + NA;

  // f32 -> bf16 conversions
  {
    int n4a = (int)(NA / 4), n4w = (int)(NW / 4);
    dim3 ga((n4a + 255) / 256), gw((n4w + 255) / 256), blk(256);
    cvt_f32_bf16<<<ga, blk, 0, stream>>>(q,  qbf, n4a);
    cvt_f32_bf16<<<ga, blk, 0, stream>>>(k,  kbf, n4a);
    cvt_f32_bf16<<<ga, blk, 0, stream>>>(v,  vbf, n4a);
    cvt_f32_bf16<<<gw, blk, 0, stream>>>(Wq, wqb, n4w);
    cvt_f32_bf16<<<gw, blk, 0, stream>>>(Wk, wkb, n4w);
    cvt_f32_bf16<<<gw, blk, 0, stream>>>(Wv, wvb, n4w);
    cvt_f32_bf16<<<gw, blk, 0, stream>>>(Wo, wob, n4w);
  }

  // QKV projections: (4096/128) x (1024/64) = 512 blocks of 8 waves
  proj_gemm_kernel<<<512, 256, 0, stream>>>(qbf, wqb, Qhb, nullptr, 0);
  proj_gemm_kernel<<<512, 256, 0, stream>>>(kbf, wkb, Khb, nullptr, 1);
  proj_gemm_kernel<<<512, 256, 0, stream>>>(vbf, wvb, Vtb, nullptr, 2);

  // Fused attention: 32 (b,h) x 16 q-blocks = 512 blocks of 8 waves
  attn_kernel<<<512, 256, 0, stream>>>(Qhb, Khb, Vtb, mask, ctxb);

  // Output projection -> f32 d_out
  proj_gemm_kernel<<<512, 256, 0, stream>>>(ctxb, wob, nullptr, out, 3);
}
